// NTMMemory_68324339745366
// MI455X (gfx1250) — matrix operand order repaired and checked by reading, not compiled, not run on hardware
//
#include <hip/hip_runtime.h>

// ---- problem constants (from reference) ----
#define B_ 1024
#define N_ 65536
#define M_ 128

typedef __attribute__((ext_vector_type(16))) _Float16 v16h;
typedef __attribute__((ext_vector_type(8)))  _Float16 v8h;
typedef __attribute__((ext_vector_type(8)))  float    v8f;
typedef __attribute__((ext_vector_type(4)))  float    v4f;

// D = A(16x32 f16) x B(32x16 f16) + C(16x16 f32), wave32 WMMA
__device__ __forceinline__ v8f wmma_f16(v16h a, v16h b, v8f c) {
    return __builtin_amdgcn_wmma_f32_16x16x32_f16(false, a, false, b,
                                                  (short)0, c, false, false);
}

// ---------------------------------------------------------------------------
// k0: transpose+convert erase/add [B,M] f32 -> eraseT/addT [M,B] f16
// ---------------------------------------------------------------------------
__global__ __launch_bounds__(256) void k0_prep(const float* __restrict__ erase,
                                               const float* __restrict__ add,
                                               _Float16* __restrict__ eraseT,
                                               _Float16* __restrict__ addT) {
    int idx = blockIdx.x * 256 + threadIdx.x;  // over B_*M_
    if (idx < B_ * M_) {
        int b = idx >> 7;         // /M_
        int m = idx & (M_ - 1);
        eraseT[m * B_ + b] = (_Float16)erase[idx];
        addT[m * B_ + b]   = (_Float16)add[idx];
    }
}

// ---------------------------------------------------------------------------
// k1: write phase. Per WG: n-tile of 128 rows, all M=128 cols.
//   erase_v/add_v[n,m] = sum_b address[b,n] * {erase,add}[b,m]   (K = B_ = 1024)
//   nc = content*(1-erase_v)+add_v  -> stored TRANSPOSED as ncT[M,N] f16.
// 8 waves; wave w owns n in [n0+16w, n0+16w+16), 8 m-tiles each,
// 16 WMMAs per 32-wide k-step per wave.
// ---------------------------------------------------------------------------
__global__ __launch_bounds__(256) void k1_write(const float* __restrict__ addr,
                                                const float* __restrict__ content,
                                                const _Float16* __restrict__ eraseT,
                                                const _Float16* __restrict__ addT,
                                                _Float16* __restrict__ ncT) {
    __shared__ _Float16 ldsNC[128 * 136];  // [m][n_rel], stride 136 (16B-aligned rows)

    const int t    = threadIdx.x;
    const int w    = t >> 5;        // wave 0..7
    const int lane = t & 31;
    const int h    = lane >> 4;     // half-wave
    const int q    = lane & 15;     // A-row / B-col / C-col index
    const int n0   = blockIdx.x * 128;

    v8f accE[8] = {}, accA[8] = {};

    for (int k0 = 0; k0 < B_; k0 += 32) {
        // A fragment: A[n_rel, k=b] = address[k0+b_rel][gn]; lane row = q.
        // element e -> b_rel = 16h+e
        const float* ap = addr + (size_t)k0 * N_ + (n0 + 16 * w + q);
        v16h aA;
#pragma unroll
        for (int e = 0; e < 16; ++e) {
            int br = 16 * h + e;
            aA[e] = (_Float16)ap[(size_t)br * N_];
        }
#pragma unroll
        for (int i = 0; i < 8; ++i) {  // m-tiles
            // B fragment: col m = 16i+q, element e -> k = 16h+e (K-contiguous in *T)
            const _Float16* ep = eraseT + (size_t)(16 * i + q) * B_ + k0 + 16 * h;
            const _Float16* dp = addT   + (size_t)(16 * i + q) * B_ + k0 + 16 * h;
            v8h e0 = *(const v8h*)ep, e1 = *(const v8h*)(ep + 8);
            v8h d0 = *(const v8h*)dp, d1 = *(const v8h*)(dp + 8);
            v16h bE, bA;
#pragma unroll
            for (int x = 0; x < 8; ++x) {
                bE[x] = e0[x]; bE[x + 8] = e1[x];
                bA[x] = d0[x]; bA[x + 8] = d1[x];
            }
            accE[i] = wmma_f16(aA, bE, accE[i]);
            accA[i] = wmma_f16(aA, bA, accA[i]);
        }
    }

    // elementwise update; park transposed tile in LDS: ldsNC[m][n_rel]
#pragma unroll
    for (int i = 0; i < 8; ++i) {
        int m = 16 * i + q;            // C col
#pragma unroll
        for (int r = 0; r < 8; ++r) {  // C row = n_rel_tile
            int nr = 16 * w + 8 * h + r;
            int n  = n0 + nr;
            float c  = content[(size_t)n * M_ + m];
            float nc = c * (1.0f - accE[i][r]) + accA[i][r];
            ldsNC[m * 136 + nr] = (_Float16)nc;
        }
    }
    __syncthreads();

    // coalesced store of ncT[m][n0..n0+128): 8 lanes x 16 halfs per m-row
#pragma unroll
    for (int pass = 0; pass < 4; ++pass) {
        int m    = pass * 32 + (t >> 3);
        int nOff = (t & 7) * 16;
        v8h c0 = *(const v8h*)&ldsNC[m * 136 + nOff];
        v8h c1 = *(const v8h*)&ldsNC[m * 136 + nOff + 8];
        *(v8h*)&ncT[(size_t)m * N_ + n0 + nOff]     = c0;
        *(v8h*)&ncT[(size_t)m * N_ + n0 + nOff + 8] = c1;
    }
}

// ---------------------------------------------------------------------------
// k2: read phase. read[b,m] = sum_n address[b,n] * nc[n,m]  (K = N_ = 65536)
// Grid = 8 b-blocks x 32 split-K chunks. WG covers [128 b x 128 m];
// wave w owns b-tile row w. Partials -> workspace (deterministic reduction).
// ---------------------------------------------------------------------------
__global__ __launch_bounds__(256) void k2_read(const float* __restrict__ addr,
                                               const _Float16* __restrict__ ncT,
                                               float* __restrict__ part) {
    const int t    = threadIdx.x;
    const int w    = t >> 5;
    const int lane = t & 31;
    const int h    = lane >> 4;
    const int q    = lane & 15;
    const int bBlk = blockIdx.x & 7;
    const int s    = blockIdx.x >> 3;        // split-K id, 0..31
    const int b0   = bBlk * 128 + w * 16;    // wave's b-tile base
    const int kBeg = s * (N_ / 32);          // 2048-wide chunk

    v8f acc[8] = {};

    for (int kk = 0; kk < N_ / 32; kk += 32) {
        const int k0 = kBeg + kk;
        // A fragment: row b = b0+q; element e -> n = k0+8h+e | k0+8h+16+(e-8)
        const float* ap = addr + (size_t)(b0 + q) * N_ + k0 + 8 * h;
        __builtin_prefetch(ap + 256, 0, 0);  // global_prefetch ahead on the stream
        v4f a0 = *(const v4f*)ap;
        v4f a1 = *(const v4f*)(ap + 4);
        v4f a2 = *(const v4f*)(ap + 16);
        v4f a3 = *(const v4f*)(ap + 20);
        v16h aF;
#pragma unroll
        for (int x = 0; x < 4; ++x) {
            aF[x]      = (_Float16)a0[x];
            aF[x + 4]  = (_Float16)a1[x];
            aF[x + 8]  = (_Float16)a2[x];
            aF[x + 12] = (_Float16)a3[x];
        }
#pragma unroll
        for (int i = 0; i < 8; ++i) {  // m-tiles
            // B fragment: col m = 16i+q, element e -> k = 8h+e | 8h+16+(e-8); ncT is [M,N] -> contiguous
            const _Float16* bp = ncT + (size_t)(16 * i + q) * N_ + k0 + 8 * h;
            v8h b0v = *(const v8h*)bp, b1v = *(const v8h*)(bp + 16);
            v16h bF;
#pragma unroll
            for (int x = 0; x < 8; ++x) { bF[x] = b0v[x]; bF[x + 8] = b1v[x]; }
            acc[i] = wmma_f16(aF, bF, acc[i]);
        }
    }

    // coalesced partial stores: part[s][b][m]
    float* pp = part + (size_t)s * (B_ * M_);
#pragma unroll
    for (int i = 0; i < 8; ++i)
#pragma unroll
        for (int r = 0; r < 8; ++r)
            pp[(size_t)(b0 + 8 * h + r) * M_ + 16 * i + q] = acc[i][r];
}

// ---------------------------------------------------------------------------
// k3: deterministic 32-way split-K reduction -> d_out
// ---------------------------------------------------------------------------
__global__ __launch_bounds__(256) void k3_reduce(const float* __restrict__ part,
                                                 float* __restrict__ out) {
    int idx = blockIdx.x * 256 + threadIdx.x;  // over B_*M_
    if (idx < B_ * M_) {
        float s = 0.0f;
#pragma unroll
        for (int i = 0; i < 32; ++i) s += part[(size_t)i * (B_ * M_) + idx];
        out[idx] = s;
    }
}

// ---------------------------------------------------------------------------
// launch: inputs in dict order: address[B,N], erase[B,M], add[B,M], content[N,M]
// workspace: [0,256K) eraseT | [256K,512K) addT | [1M,17M) ncT | [17M,33M) partials
// ---------------------------------------------------------------------------
extern "C" void kernel_launch(void* const* d_in, const int* in_sizes, int n_in,
                              void* d_out, int out_size, void* d_ws, size_t ws_size,
                              hipStream_t stream) {
    const float* address = (const float*)d_in[0];
    const float* erase   = (const float*)d_in[1];
    const float* add     = (const float*)d_in[2];
    const float* content = (const float*)d_in[3];
    float* out = (float*)d_out;

    char* ws = (char*)d_ws;
    _Float16* eraseT = (_Float16*)(ws);
    _Float16* addT   = (_Float16*)(ws + (256u << 10));
    _Float16* ncT    = (_Float16*)(ws + (1u << 20));
    float*    part   = (float*)(ws + (1u << 20) + (size_t)N_ * M_ * sizeof(_Float16));
    // needs ~33 MB of workspace

    k0_prep <<<(B_ * M_ + 255) / 256, 256, 0, stream>>>(erase, add, eraseT, addT);
    k1_write<<<N_ / 128,             256, 0, stream>>>(address, content, eraseT, addT, ncT);
    k2_read <<<8 * 32,               256, 0, stream>>>(address, ncT, part);
    k3_reduce<<<(B_ * M_ + 255) / 256, 256, 0, stream>>>(part, out);
}
